// EvaAttention_76124000354588
// MI455X (gfx1250) — matrix operand order repaired and checked by reading, not compiled, not run on hardware
//
#include <hip/hip_runtime.h>

// ---------------------------------------------------------------------------
// EVA attention block for gfx1250 (MI455X): bf16 WMMA everywhere GEMM-shaped.
// ---------------------------------------------------------------------------

typedef __attribute__((ext_vector_type(16))) __bf16 v16bf;
typedef __attribute__((ext_vector_type(8)))  __bf16 v8bf;
typedef __attribute__((ext_vector_type(8)))  float  v8f;

constexpr int   kB    = 8;
constexpr int   kN    = 1025;
constexpr int   kC    = 768;
constexpr int   kH    = 12;
constexpr int   kD    = 64;
constexpr int   MTOK  = kB * kN;     // 8200 token rows
constexpr int   NPAD  = 1056;        // padded key count for V^T (>= 1024+32)
constexpr int   QTILES  = (kN + 15) / 16;   // 65
constexpr int   QGROUPS = (QTILES + 3) / 4; // 17
constexpr float SCALE = 0.125f;      // 64^-0.5

__device__ __forceinline__ v8f wmma_bf16(v16bf a, v16bf b, v8f c) {
  // v_wmma_f32_16x16x32_bf16
  return __builtin_amdgcn_wmma_f32_16x16x32_bf16(false, a, false, b,
                                                 (short)0, c, false, false);
}

// --------------------------- small prep kernels ----------------------------

__global__ void k_cvt_bf16(const float* __restrict__ in, __bf16* __restrict__ out, int n) {
  int i = blockIdx.x * blockDim.x + threadIdx.x;
  if (i < n) out[i] = (__bf16)in[i];
}

// in: [K, Nc] row-major f32  ->  out: [Nc, K] row-major bf16
__global__ void k_transpose_bf16(const float* __restrict__ in, __bf16* __restrict__ out,
                                 int K, int Nc) {
  int i = blockIdx.x * blockDim.x + threadIdx.x;
  if (i >= K * Nc) return;
  int n = i % Nc, k = i / Nc;
  out[(size_t)n * K + k] = (__bf16)in[i];
}

__global__ void k_zero_bf16(__bf16* __restrict__ out, int n) {
  int i = blockIdx.x * blockDim.x + threadIdx.x;
  if (i < n) out[i] = (__bf16)0.0f;
}

// ------------------------------ QKV GEMM -----------------------------------
// A: x_bf16 [MTOK, 768]; BT: w_qkv^T bf16 [2304, 768].
// Each wave: one 16-row tile x 64 cols (4 WMMA accumulators), K in steps of 32.
// Epilogue: add qkv bias, scatter to Q/K [B,H,N,64] and V^T [B,H,64,NPAD].

__global__ __launch_bounds__(128) void k_qkv_gemm(
    const __bf16* __restrict__ A, const __bf16* __restrict__ BT,
    const float* __restrict__ qb, const float* __restrict__ vb,
    __bf16* __restrict__ Qo, __bf16* __restrict__ Ko, __bf16* __restrict__ Vt)
{
  const int wid  = threadIdx.x >> 5;
  const int lane = threadIdx.x & 31;
  const int hi = lane >> 4, ln = lane & 15;
  const int mt = blockIdx.x;
  const int ncol0 = (blockIdx.y * 4 + wid) * 64;
  const int row  = mt * 16 + ln;
  const int rowc = row < MTOK ? row : MTOK - 1;

  v8f acc[4] = {};
  for (int kc = 0; kc < kC; kc += 32) {
    // A fragment (16x32 bf16): elems 0..15 = K hi*16..+15 (contiguous 16)
    v8bf a0 = *(const v8bf*)(A + (size_t)rowc * kC + kc + hi * 16);
    v8bf a1 = *(const v8bf*)(A + (size_t)rowc * kC + kc + hi * 16 + 8);
    v16bf af;
#pragma unroll
    for (int i = 0; i < 8; ++i) { af[i] = a0[i]; af[8 + i] = a1[i]; }
#pragma unroll
    for (int t = 0; t < 4; ++t) {
      // B fragment (32x16 bf16): col = ln, K = hi*16 + e (contiguous 16)
      int col = ncol0 + t * 16 + ln;
      v16bf bfrag = *(const v16bf*)(BT + (size_t)col * kC + kc + hi * 16);
      acc[t] = wmma_bf16(af, bfrag, acc[t]);
    }
  }
#pragma unroll
  for (int t = 0; t < 4; ++t) {
    const int col  = ncol0 + t * 16 + ln;
    const int sect = col / kC;            // 0=q, 1=k, 2=v
    const int cc   = col - sect * kC;
    const int h = cc >> 6, d = cc & 63;
    const float bias = (sect == 0) ? qb[cc] : (sect == 2 ? vb[cc] : 0.0f);
#pragma unroll
    for (int j = 0; j < 8; ++j) {         // C/D layout: row = j + hi*8
      int r = mt * 16 + hi * 8 + j;
      if (r >= MTOK) continue;
      float v = acc[t][j] + bias;
      int b = r / kN, n = r - b * kN;
      size_t bh = (size_t)(b * kH + h);
      if (sect == 0)      Qo[(bh * kN + n) * kD + d] = (__bf16)v;
      else if (sect == 1) Ko[(bh * kN + n) * kD + d] = (__bf16)v;
      else                Vt[(bh * kD + d) * NPAD + n] = (__bf16)v;
    }
  }
}

// ------------------------------- RoPE --------------------------------------
// In-place on Q/K bf16 [B*H, N, 64]; q gets the 1/sqrt(D) scale folded in.

__global__ void k_rope(__bf16* __restrict__ Q, __bf16* __restrict__ Kb,
                       const float* __restrict__ rope, int n) {
  int idx = blockIdx.x * blockDim.x + threadIdx.x;
  if (idx >= n) return;
  int i    = idx & 31;                // pair index, d = 2i
  int rest = idx >> 5;
  int nt   = rest % kN;
  int bh   = rest / kN;
  size_t base = ((size_t)bh * kN + nt) * kD + 2 * i;
  float q0 = (float)Q[base], q1 = (float)Q[base + 1];
  if (nt == 0) {                      // CLS token: scale only
    Q[base]     = (__bf16)(q0 * SCALE);
    Q[base + 1] = (__bf16)(q1 * SCALE);
    return;
  }
  const float* rp = rope + (size_t)(nt - 1) * (2 * kD);
  float s0 = rp[2 * i], s1 = rp[2 * i + 1];
  float c0 = rp[kD + 2 * i], c1 = rp[kD + 2 * i + 1];
  float k0 = (float)Kb[base], k1 = (float)Kb[base + 1];
  Q[base]      = (__bf16)((q0 * c0 - q1 * s0) * SCALE);
  Q[base + 1]  = (__bf16)((q1 * c1 + q0 * s1) * SCALE);
  Kb[base]     = (__bf16)(k0 * c0 - k1 * s0);
  Kb[base + 1] = (__bf16)(k1 * c1 + k0 * s1);
}

// ---------------------------- Attention ------------------------------------
// One wave per 16-query tile; flash-style online softmax over 32-key blocks.
// S via WMMA(Q, K^T), P re-laid C->A through per-wave LDS, O += WMMA(P, V^T).

__global__ __launch_bounds__(128) void k_attn(
    const __bf16* __restrict__ Q, const __bf16* __restrict__ K,
    const __bf16* __restrict__ Vt, __bf16* __restrict__ Y)
{
  __shared__ float Ps[4][16][32];       // per-wave P staging (no cross-wave use)
  const int wid  = threadIdx.x >> 5;
  const int lane = threadIdx.x & 31;
  const int hi = lane >> 4, ln = lane & 15;
  const int bh = blockIdx.x / QGROUPS;
  const int qt = (blockIdx.x % QGROUPS) * 4 + wid;
  if (qt >= QTILES) return;             // uniform per wave

  const __bf16* Qb = Q  + (size_t)bh * kN * kD;
  const __bf16* Kb = K  + (size_t)bh * kN * kD;
  const __bf16* Vb = Vt + (size_t)bh * kD * NPAD;

  const int qrow = qt * 16 + ln;
  const int qrc  = qrow < kN ? qrow : kN - 1;
  v16bf qf[2];
#pragma unroll
  for (int c = 0; c < 2; ++c) {
    v8bf lo = *(const v8bf*)(Qb + (size_t)qrc * kD + c * 32 + hi * 16);
    v8bf hh = *(const v8bf*)(Qb + (size_t)qrc * kD + c * 32 + hi * 16 + 8);
#pragma unroll
    for (int i = 0; i < 8; ++i) { qf[c][i] = lo[i]; qf[c][8 + i] = hh[i]; }
  }

  v8f O[4] = {};
  float mrow[8], lrow[8];
#pragma unroll
  for (int j = 0; j < 8; ++j) { mrow[j] = -__builtin_inff(); lrow[j] = 0.0f; }

  for (int kb = 0; kb < kN; kb += 32) {
    // ---- S = Q K^T for 32 keys (two 16x16 tiles) ----
    v8f s0 = {}, s1 = {};
    const int key0 = kb + ln,      k0c = key0 < kN ? key0 : kN - 1;
    const int key1 = kb + 16 + ln, k1c = key1 < kN ? key1 : kN - 1;
#pragma unroll
    for (int c = 0; c < 2; ++c) {
      v16bf kf0 = *(const v16bf*)(Kb + (size_t)k0c * kD + c * 32 + hi * 16);
      s0 = wmma_bf16(qf[c], kf0, s0);
      v16bf kf1 = *(const v16bf*)(Kb + (size_t)k1c * kD + c * 32 + hi * 16);
      s1 = wmma_bf16(qf[c], kf1, s1);
    }
    if (key0 >= kN) {
#pragma unroll
      for (int j = 0; j < 8; ++j) s0[j] = -__builtin_inff();
    }
    if (key1 >= kN) {
#pragma unroll
      for (int j = 0; j < 8; ++j) s1[j] = -__builtin_inff();
    }

    // ---- online softmax stats (row m = j + hi*8 lives in one 16-lane group)
    float rmax[8];
#pragma unroll
    for (int j = 0; j < 8; ++j) rmax[j] = fmaxf(s0[j], s1[j]);
#pragma unroll
    for (int off = 1; off < 16; off <<= 1)
#pragma unroll
      for (int j = 0; j < 8; ++j)
        rmax[j] = fmaxf(rmax[j], __shfl_xor(rmax[j], off, 32));

    float alpha[8];
#pragma unroll
    for (int j = 0; j < 8; ++j) {
      float mn = fmaxf(mrow[j], rmax[j]);
      alpha[j] = __expf(mrow[j] - mn);
      mrow[j]  = mn;
    }
    float rsum[8];
#pragma unroll
    for (int j = 0; j < 8; ++j) {
      s0[j] = __expf(s0[j] - mrow[j]);
      s1[j] = __expf(s1[j] - mrow[j]);
      rsum[j] = s0[j] + s1[j];
    }
#pragma unroll
    for (int off = 1; off < 16; off <<= 1)
#pragma unroll
      for (int j = 0; j < 8; ++j) rsum[j] += __shfl_xor(rsum[j], off, 32);
#pragma unroll
    for (int j = 0; j < 8; ++j) lrow[j] = lrow[j] * alpha[j] + rsum[j];
#pragma unroll
    for (int t = 0; t < 4; ++t)
#pragma unroll
      for (int j = 0; j < 8; ++j) O[t][j] *= alpha[j];

    // ---- stage P (C layout) to LDS, reload as A-fragment bf16 ----
#pragma unroll
    for (int j = 0; j < 8; ++j) {
      Ps[wid][hi * 8 + j][ln]      = s0[j];
      Ps[wid][hi * 8 + j][16 + ln] = s1[j];
    }
    v16bf pf;
#pragma unroll
    for (int i = 0; i < 8; ++i) {
      pf[i]     = (__bf16)Ps[wid][ln][hi * 16 + i];
      pf[8 + i] = (__bf16)Ps[wid][ln][hi * 16 + 8 + i];
    }

    // ---- O += P V  (V^T rows contiguous in key -> clean B fragments) ----
#pragma unroll
    for (int t = 0; t < 4; ++t) {
      v16bf vf = *(const v16bf*)(Vb + (size_t)(t * 16 + ln) * NPAD + kb + hi * 16);
      O[t] = wmma_bf16(pf, vf, O[t]);
    }
  }

  const int b = bh / kH, h = bh - b * kH;
#pragma unroll
  for (int t = 0; t < 4; ++t)
#pragma unroll
    for (int j = 0; j < 8; ++j) {
      int tok = qt * 16 + hi * 8 + j;
      if (tok < kN) {
        float v = O[t][j] / lrow[j];
        Y[((size_t)(b * kN + tok)) * kC + h * kD + t * 16 + ln] = (__bf16)v;
      }
    }
}

// ------------------------------ LayerNorm ----------------------------------

__device__ __forceinline__ float block_sum(float v) {
#pragma unroll
  for (int off = 16; off > 0; off >>= 1) v += __shfl_xor(v, off, 32);
  __shared__ float sh[8];
  int wid = threadIdx.x >> 5, lane = threadIdx.x & 31;
  __syncthreads();
  if (lane == 0) sh[wid] = v;
  __syncthreads();
  if (threadIdx.x == 0) {
    float s = 0.0f;
    for (int i = 0; i < 8; ++i) s += sh[i];
    sh[0] = s;
  }
  __syncthreads();
  return sh[0];
}

__global__ __launch_bounds__(256) void k_layernorm(
    const __bf16* __restrict__ Y, const float* __restrict__ gamma,
    const float* __restrict__ beta, __bf16* __restrict__ Yln)
{
  const int row = blockIdx.x;
  const __bf16* yr = Y + (size_t)row * kC;
  float vals[3], s = 0.0f;
#pragma unroll
  for (int i = 0; i < 3; ++i) { vals[i] = (float)yr[threadIdx.x + i * 256]; s += vals[i]; }
  float mu = block_sum(s) * (1.0f / kC);
  float vs = 0.0f;
#pragma unroll
  for (int i = 0; i < 3; ++i) { float d = vals[i] - mu; vs += d * d; }
  float inv = rsqrtf(block_sum(vs) * (1.0f / kC) + 1e-6f);
#pragma unroll
  for (int i = 0; i < 3; ++i) {
    int c = threadIdx.x + i * 256;
    Yln[(size_t)row * kC + c] = (__bf16)((vals[i] - mu) * inv * gamma[c] + beta[c]);
  }
}

// --------------------------- Projection GEMM -------------------------------

__global__ __launch_bounds__(128) void k_proj_gemm(
    const __bf16* __restrict__ A, const __bf16* __restrict__ BT,
    const float* __restrict__ bias, float* __restrict__ out)
{
  const int wid  = threadIdx.x >> 5;
  const int lane = threadIdx.x & 31;
  const int hi = lane >> 4, ln = lane & 15;
  const int mt = blockIdx.x;
  const int ncol0 = (blockIdx.y * 4 + wid) * 64;
  const int row  = mt * 16 + ln;
  const int rowc = row < MTOK ? row : MTOK - 1;

  v8f acc[4] = {};
  for (int kc = 0; kc < kC; kc += 32) {
    v8bf a0 = *(const v8bf*)(A + (size_t)rowc * kC + kc + hi * 16);
    v8bf a1 = *(const v8bf*)(A + (size_t)rowc * kC + kc + hi * 16 + 8);
    v16bf af;
#pragma unroll
    for (int i = 0; i < 8; ++i) { af[i] = a0[i]; af[8 + i] = a1[i]; }
#pragma unroll
    for (int t = 0; t < 4; ++t) {
      int col = ncol0 + t * 16 + ln;
      v16bf bfrag = *(const v16bf*)(BT + (size_t)col * kC + kc + hi * 16);
      acc[t] = wmma_bf16(af, bfrag, acc[t]);
    }
  }
#pragma unroll
  for (int t = 0; t < 4; ++t) {
    int col = ncol0 + t * 16 + ln;
    float bv = bias[col];
#pragma unroll
    for (int j = 0; j < 8; ++j) {
      int r = mt * 16 + hi * 8 + j;
      if (r < MTOK) out[(size_t)r * kC + col] = acc[t][j] + bv;
    }
  }
}

// ------------------------------- launch ------------------------------------

extern "C" void kernel_launch(void* const* d_in, const int* in_sizes, int n_in,
                              void* d_out, int out_size, void* d_ws, size_t ws_size,
                              hipStream_t stream) {
  const float* x      = (const float*)d_in[0];
  const float* rope   = (const float*)d_in[1];
  const float* w_qkv  = (const float*)d_in[2];
  const float* q_bias = (const float*)d_in[3];
  const float* v_bias = (const float*)d_in[4];
  const float* gamma  = (const float*)d_in[5];
  const float* beta   = (const float*)d_in[6];
  const float* w_proj = (const float*)d_in[7];
  const float* b_proj = (const float*)d_in[8];
  float* out = (float*)d_out;
  (void)in_sizes; (void)n_in; (void)out_size; (void)ws_size;

  char* base = (char*)d_ws;
  size_t off = 0;
  auto alloc = [&](size_t elems) -> __bf16* {
    __bf16* p = (__bf16*)(base + off);
    off = (off + elems * sizeof(__bf16) + 255) & ~(size_t)255;
    return p;
  };
  __bf16* xb     = alloc((size_t)MTOK * kC);          // x in bf16
  __bf16* wqkvT  = alloc((size_t)3 * kC * kC);        // w_qkv^T
  __bf16* wprojT = alloc((size_t)kC * kC);            // w_proj^T
  __bf16* Qb     = alloc((size_t)kB * kH * kN * kD);  // Q [B,H,N,64]
  __bf16* Kb     = alloc((size_t)kB * kH * kN * kD);  // K [B,H,N,64]
  __bf16* Vt     = alloc((size_t)kB * kH * kD * NPAD);// V^T [B,H,64,NPAD]
  __bf16* Yat    = alloc((size_t)MTOK * kC);          // attention output
  __bf16* Yln    = alloc((size_t)MTOK * kC);          // layernorm output

  int n;
  n = MTOK * kC;
  k_cvt_bf16<<<(n + 255) / 256, 256, 0, stream>>>(x, xb, n);
  n = 3 * kC * kC;
  k_transpose_bf16<<<(n + 255) / 256, 256, 0, stream>>>(w_qkv, wqkvT, kC, 3 * kC);
  n = kC * kC;
  k_transpose_bf16<<<(n + 255) / 256, 256, 0, stream>>>(w_proj, wprojT, kC, kC);
  n = kB * kH * kD * NPAD;
  k_zero_bf16<<<(n + 255) / 256, 256, 0, stream>>>(Vt, n);

  dim3 g_qkv((MTOK + 15) / 16, (3 * kC) / 256);
  k_qkv_gemm<<<g_qkv, 128, 0, stream>>>(xb, wqkvT, q_bias, v_bias, Qb, Kb, Vt);

  n = kB * kH * kN * (kD / 2);
  k_rope<<<(n + 255) / 256, 256, 0, stream>>>(Qb, Kb, rope, n);

  k_attn<<<kB * kH * QGROUPS, 128, 0, stream>>>(Qb, Kb, Vt, Yat);

  k_layernorm<<<MTOK, 256, 0, stream>>>(Yat, gamma, beta, Yln);

  dim3 g_proj((MTOK + 15) / 16, kC / 256);
  k_proj_gemm<<<g_proj, 128, 0, stream>>>(Yln, wprojT, b_proj, out);
}
